// CrossAttn_58557584114423
// MI455X (gfx1250) — matrix operand order repaired and checked by reading, not compiled, not run on hardware
//
#include <hip/hip_runtime.h>
#include <hip/hip_bf16.h>

// ---------------------------------------------------------------------------
// Cross-attention on gfx1250 via bf16 WMMA (V_WMMA_F32_16X16X32_BF16).
// q = (x@Wq^T+bq)*scale ; k,v = ctx@W{k,v}^T+b ; flash-softmax(q k^T) v ; @Wo^T+bo
// CDNA5 features: v_wmma bf16, DPP ROW_XMASK reductions, async global->LDS (ASYNCcnt)
// ---------------------------------------------------------------------------

typedef __attribute__((ext_vector_type(16))) __bf16 v16bf;
typedef __attribute__((ext_vector_type(8)))  __bf16 v8bf;
typedef __attribute__((ext_vector_type(4)))  __bf16 v4bf;
typedef __attribute__((ext_vector_type(8)))  float  v8f;

#define WMMA_BF16(a, b, c) \
    __builtin_amdgcn_wmma_f32_16x16x32_bf16(false, (a), false, (b), (short)0, (c), false, false)

static __device__ __forceinline__ v16bf combine16(v8bf lo, v8bf hi) {
    return __builtin_shufflevector(lo, hi, 0,1,2,3,4,5,6,7,8,9,10,11,12,13,14,15);
}

// A-fragment (16x32 bf16, M x K): lane holds row m = lane&15.
// lanes 0-15 : elems 0-7 -> K 0..7,  elems 8-15 -> K 16..23
// lanes 16-31: elems 0-7 -> K 8..15, elems 8-15 -> K 24..31
static __device__ __forceinline__ v16bf load_a_frag(const __bf16* base, int ld, int kbase) {
    const int lane = threadIdx.x & 31;
    const __bf16* p = base + (size_t)(lane & 15) * ld + kbase + ((lane >> 4) << 3);
    v8bf lo = *(const v8bf*)(p);
    v8bf hi = *(const v8bf*)(p + 16);
    return combine16(lo, hi);
}

// B-fragment (32x16 bf16, K x N) from row-major [N][K] memory (stride ld):
// lane holds col n = lane&15; lanes 0-15: K 0..7,16..23, lanes 16-31: K 8..15,24..31
static __device__ __forceinline__ v16bf load_b_frag(const __bf16* base, int ld, int kbase) {
    const int lane = threadIdx.x & 31;
    const __bf16* p = base + (size_t)(lane & 15) * ld + kbase + ((lane >> 4) << 3);
    v8bf lo = *(const v8bf*)(p);
    v8bf hi = *(const v8bf*)(p + 16);
    return combine16(lo, hi);
}

// Butterfly max across each 16-lane half via DPP ROW_XMASK (VALU, no LDS traffic)
static __device__ __forceinline__ float rowmax16(float x) {
    float y;
    y = __int_as_float(__builtin_amdgcn_update_dpp(0, __float_as_int(x), 0x161, 0xf, 0xf, true));
    x = fmaxf(x, y);
    y = __int_as_float(__builtin_amdgcn_update_dpp(0, __float_as_int(x), 0x162, 0xf, 0xf, true));
    x = fmaxf(x, y);
    y = __int_as_float(__builtin_amdgcn_update_dpp(0, __float_as_int(x), 0x164, 0xf, 0xf, true));
    x = fmaxf(x, y);
    y = __int_as_float(__builtin_amdgcn_update_dpp(0, __float_as_int(x), 0x168, 0xf, 0xf, true));
    x = fmaxf(x, y);
    return x;
}

// ---------------------------------------------------------------------------
// fp32 -> bf16 conversion (4 elems/thread)
// ---------------------------------------------------------------------------
__global__ __launch_bounds__(256) void cvt_kernel(const float* __restrict__ in,
                                                  __bf16* __restrict__ out, int n) {
    int i = (blockIdx.x * 256 + threadIdx.x) * 4;
    if (i + 3 < n) {
        float4 f = *(const float4*)(in + i);
        v4bf o;
        o.x = (__bf16)f.x; o.y = (__bf16)f.y; o.z = (__bf16)f.z; o.w = (__bf16)f.w;
        *(v4bf*)(out + i) = o;
    }
}

// ---------------------------------------------------------------------------
// C[M,N] = (A[M,K] @ W[N,K]^T + bias) * out_scale
// block tile 128x128 (8 waves), wave tile 32x64
// ---------------------------------------------------------------------------
template <bool OUT_F32>
__global__ __launch_bounds__(256) void gemm_bias_kernel(
    const __bf16* __restrict__ A, const __bf16* __restrict__ W,
    const float* __restrict__ bias, float* __restrict__ outF,
    __bf16* __restrict__ outB, int M, int N, int K, float out_scale) {
    const int lane = threadIdx.x & 31;
    const int wid  = threadIdx.x >> 5;
    const int wm = blockIdx.y * 128 + (wid & 3) * 32;   // wave row base
    const int wn = blockIdx.x * 128 + (wid >> 2) * 64;  // wave col base

    v8f acc[2][4];
#pragma unroll
    for (int mt = 0; mt < 2; ++mt)
#pragma unroll
        for (int nt = 0; nt < 4; ++nt) acc[mt][nt] = (v8f){0.f,0.f,0.f,0.f,0.f,0.f,0.f,0.f};

    for (int k0 = 0; k0 < K; k0 += 32) {
        v16bf a[2];
        a[0] = load_a_frag(A + (size_t)wm * K, K, k0);
        a[1] = load_a_frag(A + (size_t)(wm + 16) * K, K, k0);
        v16bf b[4];
#pragma unroll
        for (int nt = 0; nt < 4; ++nt)
            b[nt] = load_b_frag(W + (size_t)(wn + nt * 16) * K, K, k0);
#pragma unroll
        for (int mt = 0; mt < 2; ++mt)
#pragma unroll
            for (int nt = 0; nt < 4; ++nt)
                acc[mt][nt] = WMMA_BF16(a[mt], b[nt], acc[mt][nt]);
    }

#pragma unroll
    for (int mt = 0; mt < 2; ++mt) {
#pragma unroll
        for (int nt = 0; nt < 4; ++nt) {
            const int col  = wn + nt * 16 + (lane & 15);
            const int row0 = wm + mt * 16 + ((lane >> 4) << 3);
            const float bv = bias[col];
#pragma unroll
            for (int r = 0; r < 8; ++r) {
                float val = (acc[mt][nt][r] + bv) * out_scale;
                size_t idx = (size_t)(row0 + r) * N + col;
                if (OUT_F32) outF[idx] = val;
                else         outB[idx] = (__bf16)val;
            }
        }
    }
}

// ---------------------------------------------------------------------------
// Flash attention: wave = 16 query rows of one (b,h); block = 8 waves, same (b,h)
// NQ=1024, NKV=512, H=16, D=64, model dim 1024. q is pre-scaled by 1/sqrt(64).
// V is staged via global_load_async_to_lds_b128 (ASYNCcnt) then transposed in LDS.
// Row sums computed with an extra WMMA against an all-ones B fragment.
// ---------------------------------------------------------------------------
__global__ __launch_bounds__(256) void attn_kernel(
    const __bf16* __restrict__ q, const __bf16* __restrict__ kmat,
    const __bf16* __restrict__ vmat, __bf16* __restrict__ out) {
    __shared__ __bf16 vrow[128 * 64];    // row-major async staging   (16 KB)
    __shared__ __bf16 vt[64 * 256];      // V^T: [dim][key_local]     (32 KB)
    __shared__ __bf16 pbuf[8][16 * 32];  // per-wave P re-layout      ( 8 KB)

    const int lane = threadIdx.x & 31;
    const int wid  = threadIdx.x >> 5;
    const int b    = blockIdx.x >> 7;          // / 128
    const int h    = (blockIdx.x >> 3) & 15;
    const int qblk = blockIdx.x & 7;
    const int qbase = qblk * 128 + wid * 16;

    const __bf16* qp = q    + ((size_t)(b * 1024 + qbase)) * 1024 + h * 64;
    const __bf16* kp = kmat + ((size_t)b * 512) * 1024 + h * 64;
    const __bf16* vp = vmat + ((size_t)b * 512) * 1024 + h * 64;

    const v16bf qa0 = load_a_frag(qp, 1024, 0);
    const v16bf qa1 = load_a_frag(qp, 1024, 32);

    v16bf ones;
#pragma unroll
    for (int i = 0; i < 16; ++i) ones[i] = (__bf16)1.0f;

    v8f o[4];
#pragma unroll
    for (int g = 0; g < 4; ++g) o[g] = (v8f){0.f,0.f,0.f,0.f,0.f,0.f,0.f,0.f};
    v8f lacc = (v8f){0.f,0.f,0.f,0.f,0.f,0.f,0.f,0.f};
    float m_arr[8];
#pragma unroll
    for (int r = 0; r < 8; ++r) m_arr[r] = -1e30f;

    for (int stage = 0; stage < 2; ++stage) {
        const int key0 = stage * 256;

        // ---- stage V^T (256 keys x 64 dims) into LDS, in two 128-key halves ----
        for (int half = 0; half < 2; ++half) {
            __syncthreads();  // all waves done reading vt/vrow from prior iteration
            const int t = threadIdx.x;
            // async copy: 128 keys x 64 dims row-major into vrow (4 x b128 per thread)
#pragma unroll
            for (int j = 0; j < 4; ++j) {
                int g    = t * 4 + j;        // 0..1023 : 8 groups of 8 dims per key
                int key  = g >> 3;           // 0..127
                int dimb = (g & 7) * 8;
                const __bf16* gp = vp + (size_t)(key0 + half * 128 + key) * 1024 + dimb;
                unsigned lds_off = (unsigned)(uintptr_t)(const void*)&vrow[key * 64 + dimb];
                asm volatile("global_load_async_to_lds_b128 %0, %1, off"
                             :: "v"(lds_off), "v"(gp) : "memory");
            }
            asm volatile("s_wait_asynccnt 0x0" ::: "memory");
            // transpose own groups vrow -> vt (thread-private groups: no barrier needed)
#pragma unroll
            for (int j = 0; j < 4; ++j) {
                int g    = t * 4 + j;
                int key  = g >> 3;
                int dimb = (g & 7) * 8;
                v8bf d = *(const v8bf*)&vrow[key * 64 + dimb];
#pragma unroll
                for (int e = 0; e < 8; ++e)
                    vt[(dimb + e) * 256 + half * 128 + key] = d[e];
            }
        }
        __syncthreads();

        // ---- flash loop over 8 chunks of 32 keys ----
        for (int c = 0; c < 8; ++c) {
            const int kb  = key0 + c * 32;  // global key base
            const int kbl = c * 32;         // key base within LDS stage
            v8f s0 = (v8f){0.f,0.f,0.f,0.f,0.f,0.f,0.f,0.f};
            v8f s1 = s0;
            s0 = WMMA_BF16(qa0, load_b_frag(kp + (size_t)kb * 1024, 1024, 0),  s0);
            s0 = WMMA_BF16(qa1, load_b_frag(kp + (size_t)kb * 1024, 1024, 32), s0);
            s1 = WMMA_BF16(qa0, load_b_frag(kp + (size_t)(kb + 16) * 1024, 1024, 0),  s1);
            s1 = WMMA_BF16(qa1, load_b_frag(kp + (size_t)(kb + 16) * 1024, 1024, 32), s1);

            float p0[8], p1[8], al[8];
#pragma unroll
            for (int r = 0; r < 8; ++r) {
                float cm = rowmax16(fmaxf(s0[r], s1[r]));
                float nm = fmaxf(m_arr[r], cm);
                al[r] = __expf(m_arr[r] - nm);
                m_arr[r] = nm;
                p0[r] = __expf(s0[r] - nm);
                p1[r] = __expf(s1[r] - nm);
            }
#pragma unroll
            for (int g = 0; g < 4; ++g)
#pragma unroll
                for (int r = 0; r < 8; ++r) o[g][r] *= al[r];
#pragma unroll
            for (int r = 0; r < 8; ++r) lacc[r] *= al[r];

            // P (16x32) through LDS into A-fragment layout
            __bf16* pw = &pbuf[wid][0];
#pragma unroll
            for (int r = 0; r < 8; ++r) {
                int row = r + ((lane >> 4) << 3);
                pw[row * 32 + (lane & 15)]      = (__bf16)p0[r];
                pw[row * 32 + 16 + (lane & 15)] = (__bf16)p1[r];
            }
            v16bf pa = load_a_frag(pw, 32, 0);

            // row sums: one WMMA against all-ones B fragment
            lacc = WMMA_BF16(pa, ones, lacc);
#pragma unroll
            for (int g = 0; g < 4; ++g) {
                v16bf vb = load_b_frag(&vt[(size_t)g * 16 * 256], 256, kbl);
                o[g] = WMMA_BF16(pa, vb, o[g]);
            }
        }
    }

    __bf16* op = out + ((size_t)(b * 1024 + qbase)) * 1024 + h * 64;
    float inv[8];
#pragma unroll
    for (int r = 0; r < 8; ++r) inv[r] = 1.f / lacc[r];
#pragma unroll
    for (int g = 0; g < 4; ++g)
#pragma unroll
        for (int r = 0; r < 8; ++r) {
            int row = r + ((lane >> 4) << 3);
            op[(size_t)row * 1024 + g * 16 + (lane & 15)] = (__bf16)(o[g][r] * inv[r]);
        }
}

// ---------------------------------------------------------------------------
// Host launch
// ---------------------------------------------------------------------------
extern "C" void kernel_launch(void* const* d_in, const int* in_sizes, int n_in,
                              void* d_out, int out_size, void* d_ws, size_t ws_size,
                              hipStream_t stream) {
    const float* x   = (const float*)d_in[0];
    const float* ctx = (const float*)d_in[1];
    const float* Wq  = (const float*)d_in[2];
    const float* bq  = (const float*)d_in[3];
    const float* Wk  = (const float*)d_in[4];
    const float* bk  = (const float*)d_in[5];
    const float* Wv  = (const float*)d_in[6];
    const float* bv  = (const float*)d_in[7];
    const float* Wo  = (const float*)d_in[8];
    const float* bo  = (const float*)d_in[9];
    float* out = (float*)d_out;

    const size_t N_X   = 8ull * 1024 * 1024;   // 8388608
    const size_t N_CTX = 8ull * 512 * 768;     // 3145728
    const size_t N_WQ  = 1024ull * 1024;
    const size_t N_WK  = 1024ull * 768;
    const size_t N_WV  = 1024ull * 768;
    const size_t N_WO  = 1024ull * 1024;
    const size_t N_Q   = 8ull * 1024 * 1024;
    const size_t N_K   = 8ull * 512 * 1024;
    const size_t N_V   = 8ull * 512 * 1024;
    const size_t N_AO  = 8ull * 1024 * 1024;

    __bf16* p = (__bf16*)d_ws;
    __bf16* x_bf  = p; p += N_X;
    __bf16* c_bf  = p; p += N_CTX;
    __bf16* wq_bf = p; p += N_WQ;
    __bf16* wk_bf = p; p += N_WK;
    __bf16* wv_bf = p; p += N_WV;
    __bf16* wo_bf = p; p += N_WO;
    __bf16* q_bf  = p; p += N_Q;
    __bf16* k_bf  = p; p += N_K;
    __bf16* v_bf  = p; p += N_V;
    __bf16* ao_bf = p; p += N_AO;

    auto cvt = [&](const float* src, __bf16* dst, size_t n) {
        cvt_kernel<<<(unsigned)(n / 1024), 256, 0, stream>>>(src, dst, (int)n);
    };
    cvt(x,   x_bf,  N_X);
    cvt(ctx, c_bf,  N_CTX);
    cvt(Wq,  wq_bf, N_WQ);
    cvt(Wk,  wk_bf, N_WK);
    cvt(Wv,  wv_bf, N_WV);
    cvt(Wo,  wo_bf, N_WO);

    // Q projection (scale 1/sqrt(64) folded in): [8192,1024]
    gemm_bias_kernel<false><<<dim3(8, 64), 256, 0, stream>>>(
        x_bf, wq_bf, bq, nullptr, q_bf, 8192, 1024, 1024, 0.125f);
    // K projection: [4096,1024] = ctx[4096,768] @ Wk^T
    gemm_bias_kernel<false><<<dim3(8, 32), 256, 0, stream>>>(
        c_bf, wk_bf, bk, nullptr, k_bf, 4096, 1024, 768, 1.0f);
    // V projection
    gemm_bias_kernel<false><<<dim3(8, 32), 256, 0, stream>>>(
        c_bf, wv_bf, bv, nullptr, v_bf, 4096, 1024, 768, 1.0f);

    // Attention: 8 batches * 16 heads * 8 q-blocks = 1024 blocks
    attn_kernel<<<1024, 256, 0, stream>>>(q_bf, k_bf, v_bf, ao_bf);

    // Output projection -> f32 d_out
    gemm_bias_kernel<true><<<dim3(8, 64), 256, 0, stream>>>(
        ao_bf, wo_bf, bo, out, nullptr, 8192, 1024, 1024, 1.0f);
}
